// MemSet_28741921145271
// MI455X (gfx1250) — compile-verified, matrix operand order checked by reading, not run on hardware
//
#include <hip/hip_runtime.h>
#include <hip/hip_bf16.h>

typedef __attribute__((ext_vector_type(16))) _Float16     v16h;
typedef __attribute__((ext_vector_type(8)))  float        v8f;
typedef __attribute__((ext_vector_type(8)))  unsigned int v8u;
typedef __attribute__((ext_vector_type(4)))  float        v4f;
typedef __attribute__((ext_vector_type(4)))  _Float16     v4h;

#define NUM_MEM 4096
#define HID     64
#define RPB     16          // rows of x per workgroup (WMMA M)
#define NROWS   32768       // 16 * 2048

union FragH {
    v16h v;
    _Float16 f[16];
    unsigned int u[8];
    v8u u8;
};

__device__ __forceinline__ int k0_of(int v, int h) {
    // 16-bit WMMA A/B fragment K-base for VGPR v, lane-half h (ISA 7.12.2)
    return ((v & 3) << 1) + ((v >> 2) << 4) + (h << 3);
}

// ---------------- prep kernel 1: inverse L2 norms of memories ----------------
__global__ void mem_norm_kernel(const float* __restrict__ mem,
                                float* __restrict__ invn) {
    int r = blockIdx.x * blockDim.x + threadIdx.x;
    if (r >= NUM_MEM) return;
    const float* p = mem + (size_t)r * HID;
    float s = 0.f;
#pragma unroll
    for (int k = 0; k < HID; ++k) { float v = p[k]; s += v * v; }
    invn[r] = 1.0f / fmaxf(sqrtf(s), 1e-12f);
}

// -------- prep kernel 2: pack f16 WMMA B-fragments into workspace -----------
// bn: GEMM1 B = m_n^T : [256 tiles][2 khalf][32 lanes][8 vgprs] packed uints
// bm: GEMM2 B = memories : [128 ktiles][4 ntiles][32 lanes][8 vgprs]
__global__ void pack_frags_kernel(const float* __restrict__ mem,
                                  const float* __restrict__ invn,
                                  unsigned int* __restrict__ bn,
                                  unsigned int* __restrict__ bm) {
    int idx = blockIdx.x * blockDim.x + threadIdx.x;   // 0 .. 262143
    if (idx < 131072) {
        int v    = idx & 7;
        int lane = (idx >> 3) & 31;
        int kh   = (idx >> 8) & 1;
        int t    = idx >> 9;
        int h = lane >> 4, n = lane & 15;
        int m = t * 16 + n;                // memory index (B column)
        int k = kh * 32 + k0_of(v, h);     // feature index (B row)
        float s = invn[m];
        _Float16 lo = (_Float16)(mem[(size_t)m * HID + k]     * s);
        _Float16 hi = (_Float16)(mem[(size_t)m * HID + k + 1] * s);
        bn[idx] = ((unsigned int)__builtin_bit_cast(unsigned short, hi) << 16) |
                   (unsigned int)__builtin_bit_cast(unsigned short, lo);
    } else {
        int j = idx - 131072;
        int v    = j & 7;
        int lane = (j >> 3) & 31;
        int nt   = (j >> 8) & 3;
        int kt   = j >> 10;
        int h = lane >> 4;
        int n = nt * 16 + (lane & 15);     // feature (B column)
        int k = kt * 32 + k0_of(v, h);     // memory index (B row = K dim)
        _Float16 lo = (_Float16)mem[(size_t)k * HID + n];
        _Float16 hi = (_Float16)mem[(size_t)(k + 1) * HID + n];
        bm[j] = ((unsigned int)__builtin_bit_cast(unsigned short, hi) << 16) |
                 (unsigned int)__builtin_bit_cast(unsigned short, lo);
    }
}

// ------------------------------- main kernel --------------------------------
__global__ void __launch_bounds__(256)
memnet_kernel(const float* __restrict__ x,
              const unsigned int* __restrict__ bn,
              const unsigned int* __restrict__ bm,
              float* __restrict__ outRead,
              float* __restrict__ outWeight) {
    __shared__ __attribute__((aligned(16))) _Float16 eBuf[RPB * NUM_MEM]; // 128 KB
    __shared__ __attribute__((aligned(16))) float    xrow[RPB * HID];     // 4 KB
    __shared__ __attribute__((aligned(16))) _Float16 aw[RPB * HID];       // 2 KB
    __shared__ __attribute__((aligned(16))) float    readBuf[RPB * HID];  // 4 KB
    __shared__ float rowSumSh[16], invDen[16], l1Sh[16], invL1[16];

    const int tid  = threadIdx.x;
    const int lane = tid & 31;
    const int wave = tid >> 5;
    const int h    = lane >> 4;
    const int n    = lane & 15;
    const int r0   = blockIdx.x * RPB;

    // ---- phase 0: load x rows (b128), L2-normalize, init accumulators ----
    {
        v4f xv = *reinterpret_cast<const v4f*>(x + (size_t)r0 * HID + tid * 4);
        *reinterpret_cast<v4f*>(&xrow[tid * 4])    = xv;
        *reinterpret_cast<v4f*>(&readBuf[tid * 4]) = (v4f){0.f, 0.f, 0.f, 0.f};
    }
    if (tid < 16) { rowSumSh[tid] = 0.f; l1Sh[tid] = 0.f; }
    __syncthreads();
    if (tid < 16) {
        float s = 0.f;
#pragma unroll
        for (int k = 0; k < HID; ++k) { float v = xrow[tid * HID + k]; s += v * v; }
        invDen[tid] = 1.0f / fmaxf(sqrtf(s), 1e-12f);   // temp: inv row norm
    }
    __syncthreads();
#pragma unroll
    for (int k = 0; k < 4; ++k) {
        int i = k * 256 + tid;
        aw[i] = (_Float16)(xrow[i] * invDen[i >> 6]);
    }
    __syncthreads();

    // ---- phase 1: logits via WMMA, exp(2*logit) into LDS, row sums ----
    FragH a0, a1;
#pragma unroll
    for (int v = 0; v < 8; ++v) {
        int k = k0_of(v, h);
        a0.f[2 * v]     = aw[n * HID + k];
        a0.f[2 * v + 1] = aw[n * HID + k + 1];
        a1.f[2 * v]     = aw[n * HID + k + 32];
        a1.f[2 * v + 1] = aw[n * HID + k + 33];
    }
    float rsum[8];
#pragma unroll
    for (int p = 0; p < 8; ++p) rsum[p] = 0.f;

    for (int it = 0; it < 32; ++it) {
        int t = wave * 32 + it;                   // N-tile: memories 16t..16t+15
        FragH b0, b1;
        b0.u8 = *reinterpret_cast<const v8u*>(bn + ((size_t)(t * 2 + 0) * 32 + lane) * 8);
        b1.u8 = *reinterpret_cast<const v8u*>(bn + ((size_t)(t * 2 + 1) * 32 + lane) * 8);
        v8f acc = {};
        acc = __builtin_amdgcn_wmma_f32_16x16x32_f16(false, a0.v, false, b0.v,
                                                     (short)0, acc, false, false);
        acc = __builtin_amdgcn_wmma_f32_16x16x32_f16(false, a1.v, false, b1.v,
                                                     (short)0, acc, false, false);
        int col = t * 16 + n;
#pragma unroll
        for (int p = 0; p < 8; ++p) {
            int row = p + 8 * h;
            float ev = __expf(2.0f * acc[p]);     // logits / TEMPERATURE(0.5)
            eBuf[row * NUM_MEM + col] = (_Float16)ev;
            rsum[p] += ev;
        }
    }
#pragma unroll
    for (int p = 0; p < 8; ++p) {
        float s = rsum[p];
#pragma unroll
        for (int off = 1; off < 16; off <<= 1) s += __shfl_xor(s, off, 32);
        if (n == 0) atomicAdd(&rowSumSh[p + 8 * h], s);
    }
    __syncthreads();
    if (tid < 16) invDen[tid] = 1.0f / rowSumSh[tid];
    __syncthreads();

    // ---- phase 2: softmax normalize + hard-shrink-relu + L1 (vectorized) ----
    for (int r = 0; r < RPB; ++r) {
        float idn = invDen[r];
        float acc1 = 0.f;
#pragma unroll
        for (int kk = 0; kk < 4; ++kk) {
            int i = r * NUM_MEM + kk * 1024 + tid * 4;
            v4h ev = *reinterpret_cast<v4h*>(&eBuf[i]);
            v4h wv;
#pragma unroll
            for (int q = 0; q < 4; ++q) {
                float w = (float)ev[q] * idn;
                float d = w - 0.0025f;
                float wp = d > 0.f ? d * w / (d + 1e-12f) : 0.f;
                wv[q] = (_Float16)wp;
                acc1 += wp;                        // wp >= 0, so |wp| == wp
            }
            *reinterpret_cast<v4h*>(&eBuf[i]) = wv;
        }
        atomicAdd(&l1Sh[r], acc1);
    }
    __syncthreads();
    if (tid < 16) invL1[tid] = 1.0f / fmaxf(l1Sh[tid], 1e-12f);
    __syncthreads();

    // ---- phase 3: L1 renorm, nontemporal b128 weight store, keep f16 copy ----
    for (int r = 0; r < RPB; ++r) {
        float il = invL1[r];
        size_t gbase = (size_t)(r0 + r) * NUM_MEM;
#pragma unroll
        for (int kk = 0; kk < 4; ++kk) {
            int c = kk * 1024 + tid * 4;
            int i = r * NUM_MEM + c;
            v4h wv = *reinterpret_cast<v4h*>(&eBuf[i]);
            v4f wf;
#pragma unroll
            for (int q = 0; q < 4; ++q) {
                wf[q] = (float)wv[q] * il;
                wv[q] = (_Float16)wf[q];
            }
            *reinterpret_cast<v4h*>(&eBuf[i]) = wv;
            __builtin_nontemporal_store(wf, reinterpret_cast<v4f*>(outWeight + gbase + c));
        }
    }
    __syncthreads();

    // ---- phase 4: read = W @ memories via WMMA (8 waves split K=4096) ----
    v8f accs[4] = {};
    for (int ktl = 0; ktl < 16; ++ktl) {
        int kt = wave * 16 + ktl;                 // K-tile: memories 32kt..32kt+31
        FragH a;
#pragma unroll
        for (int v = 0; v < 8; ++v) {
            int k = kt * 32 + k0_of(v, h);
            a.f[2 * v]     = eBuf[n * NUM_MEM + k];
            a.f[2 * v + 1] = eBuf[n * NUM_MEM + k + 1];
        }
#pragma unroll
        for (int ntl = 0; ntl < 4; ++ntl) {
            FragH b;
            b.u8 = *reinterpret_cast<const v8u*>(bm + ((size_t)(kt * 4 + ntl) * 32 + lane) * 8);
            accs[ntl] = __builtin_amdgcn_wmma_f32_16x16x32_f16(false, a.v, false, b.v,
                                                               (short)0, accs[ntl], false, false);
        }
    }
#pragma unroll
    for (int ntl = 0; ntl < 4; ++ntl)
#pragma unroll
        for (int p = 0; p < 8; ++p)
            atomicAdd(&readBuf[(p + 8 * h) * HID + ntl * 16 + n], accs[ntl][p]);
    __syncthreads();

    {
        v4f rv = *reinterpret_cast<const v4f*>(&readBuf[tid * 4]);
        __builtin_nontemporal_store(rv,
            reinterpret_cast<v4f*>(outRead + (size_t)r0 * HID + tid * 4));
    }
}

extern "C" void kernel_launch(void* const* d_in, const int* in_sizes, int n_in,
                              void* d_out, int out_size, void* d_ws, size_t ws_size,
                              hipStream_t stream) {
    const float* x   = (const float*)d_in[0];
    const float* mem = (const float*)d_in[1];

    float* out       = (float*)d_out;
    float* outRead   = out;                                  // [32768, 64]
    float* outWeight = out + (size_t)NROWS * HID;            // [32768, 4096]

    // workspace layout: invn (16 KB) | bn frags (512 KB) | bm frags (512 KB)
    float*        invn = (float*)d_ws;
    unsigned int* bn   = (unsigned int*)((char*)d_ws + 16384);
    unsigned int* bm   = bn + 131072;

    mem_norm_kernel<<<NUM_MEM / 256, 256, 0, stream>>>(mem, invn);
    pack_frags_kernel<<<262144 / 256, 256, 0, stream>>>(mem, invn, bn, bm);
    memnet_kernel<<<NROWS / RPB, 256, 0, stream>>>(x, bn, bm, outRead, outWeight);
}
